// MoxELayer_14345190769345
// MI455X (gfx1250) — compile-verified, matrix operand order checked by reading, not run on hardware
//
#include <hip/hip_runtime.h>

// ---------------------------------------------------------------------------
// MoE forward for MI455X (gfx1250, wave32, WMMA + async-to-LDS).
//
// Roofline: sparse MoE FLOPs = 275 GFLOP; ~600 MB moved => ~25us memory floor
// @23.3TB/s. bf16 WMMA (16x16x32, f32 accum) is ~8x the f32 WMMA (16x16x4)
// ceiling, so FFN GEMMs run bf16/f32-accum; router + losses stay f32.
// B-tiles are staged into LDS with GLOBAL_LOAD_ASYNC_TO_LDS_B128 (ASYNCcnt,
// double-buffered) so each 128x128 block fetches weights once instead of 4x.
// ---------------------------------------------------------------------------

typedef __bf16 bf16_t;
typedef bf16_t v16bf __attribute__((ext_vector_type(16)));
typedef float  v8f   __attribute__((ext_vector_type(8)));

typedef __attribute__((address_space(3))) unsigned short as3_u16;
typedef as3_u16* lds_u16p;

static constexpr int Ntok  = 8192;   // B*S
static constexpr int Dd    = 1024;
static constexpr int Ff    = 4096;
static constexpr int Ee    = 8;
static constexpr int TOPK  = 2;
static constexpr int NSLOT = Ntok * TOPK;          // 16384
static constexpr int MAXTILES = NSLOT / 128 + Ee;  // 136 (expert-aligned M tiles)

__device__ __forceinline__ unsigned short f2bf(float f) {
  unsigned int u = __float_as_uint(f);
  unsigned int r = u + 0x7FFFu + ((u >> 16) & 1u);   // round-to-nearest-even
  if ((u & 0x7F800000u) == 0x7F800000u) r = u;       // keep Inf/NaN payload
  return (unsigned short)(r >> 16);
}

__device__ __forceinline__ float gelu_tanh(float x) {
  const float c = 0.7978845608028654f;               // sqrt(2/pi)
  float t = tanhf(c * (x + 0.044715f * x * x * x));
  return 0.5f * x * (1.0f + t);
}

// Load one 16-lane x 32-K bf16 fragment half for this lane.
// ISA layout (CDNA5 ISA): lanes 0-15 hold K {0..7,16..23}, lanes 16-31 hold
// K {8..15,24..31}. Caller pre-biases the pointer by 8*(lane>>4) elements;
// we read 16B at +0 and 16B at +32B. Works for global or LDS (generic ptr).
__device__ __forceinline__ v16bf load_frag(const unsigned short* p16) {
  union { uint4 q[2]; v16bf v; } u;
  const uint4* p = (const uint4*)p16;
  u.q[0] = p[0];
  u.q[1] = p[2];
  return u.v;
}

__device__ __forceinline__ v8f wmma_bf16(v16bf a, v16bf b, v8f c) {
  return __builtin_amdgcn_wmma_f32_16x16x32_bf16(
      false, a, false, b, (short)0, c, false, false);
}

// gfx1250 async copy: 16B global -> LDS, tracked by ASYNCcnt (07_vmem §15.18,
// 08_async_tensor §4). VDST = LDS byte offset VGPR, VADDR = 64-bit address.
// Inline asm (not builtin) for toolchain portability per CDNA5_HIP.md.
__device__ __forceinline__ void async_g2l_16B(lds_u16p l, const unsigned short* g) {
  asm volatile("global_load_async_to_lds_b128 %0, %1, off"
               :: "v"(l), "v"(g) : "memory");
}
__device__ __forceinline__ void wait_async_le2() {
#if __has_builtin(__builtin_amdgcn_s_wait_asynccnt)
  __builtin_amdgcn_s_wait_asynccnt(2);
#else
  asm volatile("s_wait_asynccnt 0x2" ::: "memory");
#endif
}
__device__ __forceinline__ void wait_async_le0() {
#if __has_builtin(__builtin_amdgcn_s_wait_asynccnt)
  __builtin_amdgcn_s_wait_asynccnt(0);
#else
  asm volatile("s_wait_asynccnt 0x0" ::: "memory");
#endif
}

// ---------------------------------------------------------------------------
// Utility kernels
// ---------------------------------------------------------------------------
__global__ __launch_bounds__(256) void zero_kernel(float* out_hidden,
                                                   int* counts, int* cursors) {
  size_t i = (size_t)blockIdx.x * blockDim.x + threadIdx.x;
  size_t stride = (size_t)gridDim.x * blockDim.x;
  const size_t HID = (size_t)Ntok * Dd;
  for (size_t j = i; j < HID; j += stride) out_hidden[j] = 0.0f;
  if (i < Ee) { counts[i] = 0; cursors[i] = 0; }
}

__global__ __launch_bounds__(256) void convert_h_kernel(const float* __restrict__ in,
                                                        unsigned short* __restrict__ out,
                                                        size_t n) {
  size_t i = (size_t)blockIdx.x * blockDim.x + threadIdx.x;
  size_t stride = (size_t)gridDim.x * blockDim.x;
  for (size_t j = i; j < n; j += stride) out[j] = f2bf(in[j]);
}

// Transpose [R,C] f32 -> [C,R] bf16, per expert (blockIdx.z), via LDS tile.
// Gives B fragments contiguous K for the WMMA per-lane layout.
__global__ __launch_bounds__(256) void transpose_bf16_kernel(const float* __restrict__ in,
                                                             unsigned short* __restrict__ out,
                                                             int R, int C) {
  __shared__ float tile[32][33];
  const size_t plane = (size_t)R * C;
  const float* src = in + (size_t)blockIdx.z * plane;
  unsigned short* dst = out + (size_t)blockIdx.z * plane;
  int c0 = blockIdx.x * 32, r0 = blockIdx.y * 32;
  int tx = threadIdx.x, ty = threadIdx.y;   // (32,8)
  for (int i = 0; i < 32; i += 8)
    tile[ty + i][tx] = src[(size_t)(r0 + ty + i) * C + c0 + tx];
  __syncthreads();
  for (int i = 0; i < 32; i += 8)
    dst[(size_t)(c0 + ty + i) * R + r0 + tx] = f2bf(tile[tx][ty + i]);
}

// ---------------------------------------------------------------------------
// Router: one wave32 per token. logits = h[t,:] @ Wg, softmax, top-2.
// ---------------------------------------------------------------------------
__global__ __launch_bounds__(256) void router_kernel(const float* __restrict__ h,
                                                     const float* __restrict__ Wg,
                                                     float* __restrict__ probs_out,
                                                     int* __restrict__ tok_e,
                                                     float* __restrict__ tok_w,
                                                     float* __restrict__ lse_arr,
                                                     int* __restrict__ counts) {
  const int wavei = threadIdx.x >> 5;
  const int lane  = threadIdx.x & 31;
  const int t = blockIdx.x * 8 + wavei;

  float part[Ee];
#pragma unroll
  for (int e = 0; e < Ee; ++e) part[e] = 0.0f;

  const float* hrow = h + (size_t)t * Dd;
  for (int d = lane; d < Dd; d += 32) {
    float hv = hrow[d];
    const float* wr = Wg + d * Ee;
#pragma unroll
    for (int e = 0; e < Ee; ++e) part[e] += hv * wr[e];
  }
#pragma unroll
  for (int e = 0; e < Ee; ++e)
    for (int off = 16; off > 0; off >>= 1)
      part[e] += __shfl_xor(part[e], off, 32);

  if (lane == 0) {
    float m = part[0];
#pragma unroll
    for (int e = 1; e < Ee; ++e) m = fmaxf(m, part[e]);
    float ex[Ee], s = 0.0f;
#pragma unroll
    for (int e = 0; e < Ee; ++e) { ex[e] = expf(part[e] - m); s += ex[e]; }
    float inv = 1.0f / s;
    float p[Ee];
#pragma unroll
    for (int e = 0; e < Ee; ++e) {
      p[e] = ex[e] * inv;
      probs_out[(size_t)t * Ee + e] = p[e];
    }
    lse_arr[t] = m + logf(s);
    // top-2, lowest index wins ties (matches jax.lax.top_k)
    int i1 = 0;
#pragma unroll
    for (int e = 1; e < Ee; ++e) if (p[e] > p[i1]) i1 = e;
    int i2 = -1;
    float best2 = -1.0f;
#pragma unroll
    for (int e = 0; e < Ee; ++e)
      if (e != i1 && p[e] > best2) { best2 = p[e]; i2 = e; }
    float wsum = p[i1] + p[i2];
    tok_e[2 * t + 0] = i1;  tok_w[2 * t + 0] = p[i1] / wsum;
    tok_e[2 * t + 1] = i2;  tok_w[2 * t + 1] = p[i2] / wsum;
    atomicAdd(&counts[i1], 1);
    atomicAdd(&counts[i2], 1);
  }
}

// ---------------------------------------------------------------------------
// Deterministic reductions (fixed tree order), offsets, tile table.
// ---------------------------------------------------------------------------
__global__ __launch_bounds__(256) void finalize_kernel(const float* __restrict__ probs,
                                                       const float* __restrict__ lse,
                                                       const int* __restrict__ counts,
                                                       float* __restrict__ out_tail,
                                                       int* __restrict__ offsets,
                                                       int4* __restrict__ tiles,
                                                       int* __restrict__ n_tiles) {
  __shared__ float red[256];
  const int tid = threadIdx.x;

  float z = 0.0f;
  for (int t = tid; t < Ntok; t += 256) { float l = lse[t]; z += l * l; }
  red[tid] = z; __syncthreads();
  for (int s = 128; s > 0; s >>= 1) { if (tid < s) red[tid] += red[tid + s]; __syncthreads(); }
  float zsum = red[0]; __syncthreads();

  float psum[Ee];
  for (int e = 0; e < Ee; ++e) {
    float p = 0.0f;
    for (int t = tid; t < Ntok; t += 256) p += probs[(size_t)t * Ee + e];
    red[tid] = p; __syncthreads();
    for (int s = 128; s > 0; s >>= 1) { if (tid < s) red[tid] += red[tid + s]; __syncthreads(); }
    psum[e] = red[0]; __syncthreads();
  }

  if (tid == 0) {
    out_tail[0] = zsum / (float)Ntok;                       // z_loss
    float lb = 0.0f;
    for (int e = 0; e < Ee; ++e)
      lb += ((float)counts[e] / (float)(Ntok * TOPK)) * (psum[e] / (float)Ntok);
    out_tail[1] = (float)Ee * lb;                           // lb_loss
    for (int e = 0; e < Ee; ++e) out_tail[2 + e] = (float)counts[e];

    int off = 0, nt = 0;
    for (int e = 0; e < Ee; ++e) {
      offsets[e] = off;
      int c = counts[e];
      for (int s = 0; s < c; s += 128) {
        int rows = (c - s) < 128 ? (c - s) : 128;
        tiles[nt++] = make_int4(off + s, rows, e, 0);
      }
      off += c;
    }
    *n_tiles = nt;
  }
}

__global__ __launch_bounds__(256) void scatter_kernel(const int* __restrict__ tok_e,
                                                      const float* __restrict__ tok_w,
                                                      const int* __restrict__ offsets,
                                                      int* __restrict__ cursors,
                                                      int* __restrict__ slot_token,
                                                      float* __restrict__ slot_w) {
  int t = blockIdx.x * blockDim.x + threadIdx.x;
  if (t >= Ntok) return;
#pragma unroll
  for (int k = 0; k < TOPK; ++k) {
    int e = tok_e[2 * t + k];
    int pos = atomicAdd(&cursors[e], 1);
    int s = offsets[e] + pos;
    slot_token[s] = t;
    slot_w[s] = tok_w[2 * t + k];
  }
}

// ---------------------------------------------------------------------------
// GEMM1: act[slot, :] = gelu( h[token(slot), :] @ W1[e] + b1[e] )  (bf16 out)
// Block tile 128(M)x128(N), 8 waves in 4x2, wave tile 32x64 = 2x4 WMMAs.
// B tile (8KB/K-step) async-DMA'd into double-buffered LDS; A direct global.
// ---------------------------------------------------------------------------
__global__ __launch_bounds__(256) void gemm1_kernel(const unsigned short* __restrict__ h_bf,
                                                    const unsigned short* __restrict__ W1t, // [E,F,D]
                                                    const float* __restrict__ b1v,          // [E,F]
                                                    const int* __restrict__ slot_token,
                                                    const int4* __restrict__ tiles,
                                                    const int* __restrict__ n_tiles,
                                                    unsigned short* __restrict__ act) {
  __shared__ unsigned short Btile[2][128][32];   // [buf][col][k] bf16, 16KB

  const int tid = (int)blockIdx.y;
  if (tid >= *n_tiles) return;
  const int4 tl = tiles[tid];
  const int row0 = tl.x, cnt = tl.y, e = tl.z;
  const int nblk = blockIdx.x * 128;
  const int lane = threadIdx.x & 31;
  const int w = threadIdx.x >> 5;
  const int wm = w >> 1, wn = w & 1;
  const int hi = lane >> 4, l15 = lane & 15;

  // async staging: thread covers 32B (= 2 x b128) of one column's 64B K-chunk
  const int tcol  = threadIdx.x >> 1;
  const int thalf = (threadIdx.x & 1) * 16;                 // element offset
  const unsigned short* gB = W1t + ((size_t)e * Ff + nblk + tcol) * Dd + thalf;
  lds_u16p ldsB = (lds_u16p)(&Btile[0][0][0]) + tcol * 32 + thalf;

  const unsigned short* Abase[2];
#pragma unroll
  for (int sm = 0; sm < 2; ++sm) {
    int mloc = wm * 32 + sm * 16 + l15;
    int r = mloc < cnt ? mloc : 0;
    Abase[sm] = h_bf + (size_t)slot_token[row0 + r] * Dd + 8 * hi;
  }

  v8f acc[2][4];
#pragma unroll
  for (int i = 0; i < 2; ++i)
#pragma unroll
    for (int j = 0; j < 4; ++j)
#pragma unroll
      for (int r = 0; r < 8; ++r) acc[i][j][r] = 0.0f;

  // prologue: stage K-step 0 into buffer 0
  async_g2l_16B(ldsB, gB);
  async_g2l_16B(ldsB + 8, gB + 8);

  const int KSTEPS = Dd / 32;
#pragma unroll 2
  for (int ks = 0; ks < KSTEPS; ++ks) {
    const int k0 = ks * 32;
    v16bf a[2];
#pragma unroll
    for (int sm = 0; sm < 2; ++sm) a[sm] = load_frag(Abase[sm] + k0);

    if (ks + 1 < KSTEPS) {        // stage next buffer, keep its 2 ops in flight
      const unsigned short* g = gB + k0 + 32;
      lds_u16p l = ldsB + ((ks + 1) & 1) * (128 * 32);
      async_g2l_16B(l, g);
      async_g2l_16B(l + 8, g + 8);
      wait_async_le2();
    } else {
      wait_async_le0();
    }
    __syncthreads();              // all waves' async data for buf ks visible

    const unsigned short* Bb = &Btile[ks & 1][0][0];
    v16bf b[4];
#pragma unroll
    for (int sn = 0; sn < 4; ++sn)
      b[sn] = load_frag(Bb + (wn * 64 + sn * 16 + l15) * 32 + 8 * hi);
#pragma unroll
    for (int sm = 0; sm < 2; ++sm)
#pragma unroll
      for (int sn = 0; sn < 4; ++sn)
        acc[sm][sn] = wmma_bf16(a[sm], b[sn], acc[sm][sn]);
    __syncthreads();              // done reading buf ks before it is rewritten
  }

#pragma unroll
  for (int sm = 0; sm < 2; ++sm) {
    int mbase = wm * 32 + sm * 16 + hi * 8;
#pragma unroll
    for (int r = 0; r < 8; ++r) {
      int mloc = mbase + r;
      if (mloc >= cnt) continue;
      size_t row = (size_t)(row0 + mloc);
#pragma unroll
      for (int sn = 0; sn < 4; ++sn) {
        int n = nblk + wn * 64 + sn * 16 + l15;
        float v = acc[sm][sn][r] + b1v[e * Ff + n];
        act[row * Ff + n] = f2bf(gelu_tanh(v));
      }
    }
  }
}

// ---------------------------------------------------------------------------
// GEMM2: out[token] += gw * ( act[slot,:] @ W2[e] + b2[e] )
// Exactly two atomic f32 adds per output element from zero => deterministic.
// ---------------------------------------------------------------------------
__global__ __launch_bounds__(256) void gemm2_kernel(const unsigned short* __restrict__ act,
                                                    const unsigned short* __restrict__ W2t, // [E,D,F]
                                                    const float* __restrict__ b2v,          // [E,D]
                                                    const int* __restrict__ slot_token,
                                                    const float* __restrict__ slot_w,
                                                    const int4* __restrict__ tiles,
                                                    const int* __restrict__ n_tiles,
                                                    float* __restrict__ out_hidden) {
  __shared__ unsigned short Btile[2][128][32];

  const int tid = (int)blockIdx.y;
  if (tid >= *n_tiles) return;
  const int4 tl = tiles[tid];
  const int row0 = tl.x, cnt = tl.y, e = tl.z;
  const int nblk = blockIdx.x * 128;   // over D
  const int lane = threadIdx.x & 31;
  const int w = threadIdx.x >> 5;
  const int wm = w >> 1, wn = w & 1;
  const int hi = lane >> 4, l15 = lane & 15;

  const int tcol  = threadIdx.x >> 1;
  const int thalf = (threadIdx.x & 1) * 16;
  const unsigned short* gB = W2t + ((size_t)e * Dd + nblk + tcol) * Ff + thalf;
  lds_u16p ldsB = (lds_u16p)(&Btile[0][0][0]) + tcol * 32 + thalf;

  const unsigned short* Abase[2];
#pragma unroll
  for (int sm = 0; sm < 2; ++sm) {
    int mloc = wm * 32 + sm * 16 + l15;
    int r = mloc < cnt ? mloc : 0;
    Abase[sm] = act + (size_t)(row0 + r) * Ff + 8 * hi;
  }

  v8f acc[2][4];
#pragma unroll
  for (int i = 0; i < 2; ++i)
#pragma unroll
    for (int j = 0; j < 4; ++j)
#pragma unroll
      for (int r = 0; r < 8; ++r) acc[i][j][r] = 0.0f;

  async_g2l_16B(ldsB, gB);
  async_g2l_16B(ldsB + 8, gB + 8);

  const int KSTEPS = Ff / 32;
#pragma unroll 2
  for (int ks = 0; ks < KSTEPS; ++ks) {
    const int k0 = ks * 32;
    v16bf a[2];
#pragma unroll
    for (int sm = 0; sm < 2; ++sm) a[sm] = load_frag(Abase[sm] + k0);

    if (ks + 1 < KSTEPS) {
      const unsigned short* g = gB + k0 + 32;
      lds_u16p l = ldsB + ((ks + 1) & 1) * (128 * 32);
      async_g2l_16B(l, g);
      async_g2l_16B(l + 8, g + 8);
      wait_async_le2();
    } else {
      wait_async_le0();
    }
    __syncthreads();

    const unsigned short* Bb = &Btile[ks & 1][0][0];
    v16bf b[4];
#pragma unroll
    for (int sn = 0; sn < 4; ++sn)
      b[sn] = load_frag(Bb + (wn * 64 + sn * 16 + l15) * 32 + 8 * hi);
#pragma unroll
    for (int sm = 0; sm < 2; ++sm)
#pragma unroll
      for (int sn = 0; sn < 4; ++sn)
        acc[sm][sn] = wmma_bf16(a[sm], b[sn], acc[sm][sn]);
    __syncthreads();
  }

#pragma unroll
  for (int sm = 0; sm < 2; ++sm) {
    int mbase = wm * 32 + sm * 16 + hi * 8;
#pragma unroll
    for (int r = 0; r < 8; ++r) {
      int mloc = mbase + r;
      if (mloc >= cnt) continue;
      int row = row0 + mloc;
      int tok = slot_token[row];
      float gw = slot_w[row];
#pragma unroll
      for (int sn = 0; sn < 4; ++sn) {
        int n = nblk + wn * 64 + sn * 16 + l15;
        float v = (acc[sm][sn][r] + b2v[e * Dd + n]) * gw;
        atomicAdd(&out_hidden[(size_t)tok * Dd + n], v);
      }
    }
  }
}

// ---------------------------------------------------------------------------
extern "C" void kernel_launch(void* const* d_in, const int* in_sizes, int n_in,
                              void* d_out, int out_size, void* d_ws, size_t ws_size,
                              hipStream_t stream) {
  (void)in_sizes; (void)n_in; (void)out_size; (void)ws_size;
  const float* h_t = (const float*)d_in[0];
  const float* Wg  = (const float*)d_in[1];
  const float* W1  = (const float*)d_in[2];
  const float* b1  = (const float*)d_in[3];
  const float* W2  = (const float*)d_in[4];
  const float* b2  = (const float*)d_in[5];
  // d_in[6] = top_k (2 by construction)

  float* out        = (float*)d_out;
  float* out_hidden = out;                                     // [8192,1024]
  float* out_probs  = out + (size_t)Ntok * Dd;                 // [8192,8]
  float* out_tail   = out_probs + (size_t)Ntok * Ee;           // z, lb, counts[8]

  char* ws = (char*)d_ws;
  size_t off = 0;
  auto take = [&](size_t bytes) -> char* {
    char* p = ws + off;
    off = (off + bytes + 255) & ~(size_t)255;
    return p;
  };
  unsigned short* h_bf = (unsigned short*)take((size_t)Ntok * Dd * 2);
  unsigned short* W1t  = (unsigned short*)take((size_t)Ee * Ff * Dd * 2);
  unsigned short* W2t  = (unsigned short*)take((size_t)Ee * Dd * Ff * 2);
  unsigned short* actb = (unsigned short*)take((size_t)NSLOT * Ff * 2);
  int*   slot_token = (int*)take((size_t)NSLOT * 4);
  float* slot_w     = (float*)take((size_t)NSLOT * 4);
  int*   tok_e      = (int*)take((size_t)Ntok * 2 * 4);
  float* tok_w      = (float*)take((size_t)Ntok * 2 * 4);
  float* lse        = (float*)take((size_t)Ntok * 4);
  int*   counts     = (int*)take(64);
  int*   cursors    = (int*)take(64);
  int*   offsets    = (int*)take(64);
  int4*  tiles      = (int4*)take(MAXTILES * sizeof(int4));
  int*   n_tiles    = (int*)take(64);

  zero_kernel<<<2048, 256, 0, stream>>>(out_hidden, counts, cursors);
  convert_h_kernel<<<2048, 256, 0, stream>>>(h_t, h_bf, (size_t)Ntok * Dd);
  transpose_bf16_kernel<<<dim3(Ff / 32, Dd / 32, Ee), dim3(32, 8), 0, stream>>>(W1, W1t, Dd, Ff);
  transpose_bf16_kernel<<<dim3(Dd / 32, Ff / 32, Ee), dim3(32, 8), 0, stream>>>(W2, W2t, Ff, Dd);
  router_kernel<<<Ntok / 8, 256, 0, stream>>>(h_t, Wg, out_probs, tok_e, tok_w, lse, counts);
  finalize_kernel<<<1, 256, 0, stream>>>(out_probs, lse, counts, out_tail, offsets, tiles, n_tiles);
  scatter_kernel<<<Ntok / 256, 256, 0, stream>>>(tok_e, tok_w, offsets, cursors, slot_token, slot_w);
  gemm1_kernel<<<dim3(Ff / 128, MAXTILES), 256, 0, stream>>>(h_bf, W1t, b1, slot_token, tiles, n_tiles, actb);
  gemm2_kernel<<<dim3(Dd / 128, MAXTILES), 256, 0, stream>>>(actb, W2t, b2, slot_token, slot_w, tiles, n_tiles, out_hidden);
}
